// ODENet_66632122630198
// MI455X (gfx1250) — compile-verified, matrix operand order checked by reading, not run on hardware
//
#include <hip/hip_runtime.h>

typedef __attribute__((ext_vector_type(16))) __bf16 v16bf;
typedef __attribute__((ext_vector_type(8)))  float  v8f;

#define SEQ   2048
#define BATCH 512
#define HID   256

#if __has_builtin(__builtin_amdgcn_tanhf)
#define TANHF(x) __builtin_amdgcn_tanhf(x)   // hardware v_tanh_f32 (CDNA5 TRANS op)
#else
#define TANHF(x) tanhf(x)
#endif

// One workgroup per 16-row batch tile; 8 waves split the N=256 output columns
// (32 columns / wave). W2 lives in registers as WMMA B-fragments for the whole
// 2048-step scan; per step only the 16x256 bf16 activation block moves via LDS.
__global__ __launch_bounds__(256, 1)
void odenet_scan_kernel(const float* __restrict__ xg,
                        const float* __restrict__ W1,
                        const float* __restrict__ b1,
                        const float* __restrict__ W2,
                        const float* __restrict__ b2,
                        const float* __restrict__ W3,
                        const float* __restrict__ b3,
                        const float* __restrict__ state0,
                        float* __restrict__ out)
{
    // 128 B-fragments (16 n-tiles x 8 k-tiles), 1KB each = 128KB
    __shared__ __align__(64) __bf16 sW2[128 * 512];
    // 8 A-fragments (16x256 bf16 activations), viewed as packed bf16 pairs
    __shared__ __align__(64) unsigned int sA32[8 * 256];
    __shared__ float sRed[8][2][8];   // [wave][half][r] layer-3 partials
    __shared__ float sX[16];
    __shared__ float sSt[16];

    const int tid  = threadIdx.x;
    const int wv   = tid >> 5;          // wave id 0..7
    const int lane = tid & 31;
    const int b0   = blockIdx.x * 16;   // batch tile base

    // ---- one-time: pack W2 (fp32 row-major KxN) into bf16 WMMA B fragments.
    // 16-bit B fragment (32x16 KxN): lane = kmid*16 + (n%16),
    // short idx within lane's 32B = khi*8 + klow, k = khi*16 + kmid*8 + klow.
    {
        const int n = tid;              // this thread owns column n of W2
        #pragma unroll 4
        for (int k = 0; k < HID; ++k) {
            const int kt   = k >> 5;
            const int kk   = k & 31;
            const int khi  = kk >> 4;
            const int kmid = (kk >> 3) & 1;
            const int klow = kk & 7;
            const int fid  = (n >> 4) * 8 + kt;          // n-tile major
            const int fl   = kmid * 16 + (n & 15);       // fragment lane
            sW2[fid * 512 + fl * 16 + khi * 8 + klow] = (__bf16)W2[k * HID + n];
        }
    }
    if (tid < 16) {
        sSt[tid] = state0[b0 + tid];
        sX[tid]  = xg[b0 + tid];        // x_0
    }
    __syncthreads();

    // ---- hoist W2 B-fragments into registers (resident for all 2048 steps)
    v16bf bfrag[2][8];
    #pragma unroll
    for (int nt = 0; nt < 2; ++nt) {
        #pragma unroll
        for (int kt = 0; kt < 8; ++kt) {
            const int fid = (wv * 2 + nt) * 8 + kt;
            bfrag[nt][kt] = *reinterpret_cast<const v16bf*>(&sW2[fid * 512 + lane * 16]);
        }
    }

    // ---- layer-1 mapping: thread owns a column PAIR (k0,k0+1) x 8 rows,
    // so the A-fragment scatter is 8 packed ds_store_b32 per thread.
    const int c2    = tid & 127;        // column-pair index
    const int k0    = c2 * 2;
    const int mbase = (tid >> 7) * 8;   // rows mbase..mbase+7
    const float w1x0 = W1[k0],       w1x1 = W1[k0 + 1];
    const float w1y0 = W1[HID + k0], w1y1 = W1[HID + k0 + 1];
    const float b1v0 = b1[k0],       b1v1 = b1[k0 + 1];
    // packed-dword address: uint idx = kt*256 + (kmid*16+m)*8 + khi*4 + klow/2
    const int a_kt   = k0 >> 5;
    const int a_khi  = (k0 & 31) >> 4;
    const int a_kmid = (k0 >> 3) & 1;
    const int a_kl2  = (k0 & 7) >> 1;
    const int a_base = a_kt * 256 + (a_kmid * 16 + mbase) * 8 + a_khi * 4 + a_kl2;

    // layer-2/3 per-lane invariants
    const int   nc   = lane & 15;       // C-fragment column within n-tile
    const float b2v0 = b2[wv * 32 + nc];
    const float b2v1 = b2[wv * 32 + 16 + nc];
    const float w3v0 = W3[wv * 32 + nc];
    const float w3v1 = W3[wv * 32 + 16 + nc];
    const float b3v  = b3[0];

    #pragma unroll 1
    for (int t = 0; t < SEQ; ++t) {
        // layer 1: h1[m][c] = tanh(x[m]*W1[0][c] + y[m]*W1[1][c] + b1[c]),
        // two columns per thread, packed bf16 pair stores into A fragments
        #pragma unroll
        for (int i = 0; i < 8; ++i) {
            const int m = mbase + i;
            const float xv = sX[m], yv = sSt[m];
            const float v0 = TANHF(fmaf(xv, w1x0, fmaf(yv, w1y0, b1v0)));
            const float v1 = TANHF(fmaf(xv, w1x1, fmaf(yv, w1y1, b1v1)));
            union { __bf16 b[2]; unsigned int u; } pk;
            pk.b[0] = (__bf16)v0;
            pk.b[1] = (__bf16)v1;
            sA32[a_base + i * 8] = pk.u;
        }
        __syncthreads();   // A fragments ready

        // Emit ys[t] and prefetch x[t+1] NOW (sSt is still the pre-update
        // state); the global-load latency hides under the WMMA phase.
        float xnext = 0.f;
        if (tid < 16) {
            out[t * BATCH + b0 + tid] = sSt[tid];
            if (t + 1 < SEQ) xnext = xg[(t + 1) * BATCH + b0 + tid];
        }

        // layer 2: (16x256) @ (256x32) per wave. 4+4 software pipeline:
        // preload fragments 0-3, issue fragments 4-7 during the first 8
        // WMMAs. Keeps >=4 DS loads in flight ahead of use while staying
        // under the 256-VGPR boundary (no s_set_vgpr_msb traffic).
        v8f acc0, acc1;
        #pragma unroll
        for (int r = 0; r < 8; ++r) { acc0[r] = b2v0; acc1[r] = b2v1; }

        v16bf a0[4], a1[4];
        #pragma unroll
        for (int kt = 0; kt < 4; ++kt)
            a0[kt] = *reinterpret_cast<const v16bf*>(&sA32[kt * 256 + lane * 8]);
        #pragma unroll
        for (int kt = 0; kt < 4; ++kt) {
            a1[kt] = *reinterpret_cast<const v16bf*>(&sA32[(kt + 4) * 256 + lane * 8]);
            acc0 = __builtin_amdgcn_wmma_f32_16x16x32_bf16(
                false, a0[kt], false, bfrag[0][kt], (short)0, acc0, false, false);
            acc1 = __builtin_amdgcn_wmma_f32_16x16x32_bf16(
                false, a0[kt], false, bfrag[1][kt], (short)0, acc1, false, false);
        }
        #pragma unroll
        for (int kt = 0; kt < 4; ++kt) {
            acc0 = __builtin_amdgcn_wmma_f32_16x16x32_bf16(
                false, a1[kt], false, bfrag[0][kt + 4], (short)0, acc0, false, false);
            acc1 = __builtin_amdgcn_wmma_f32_16x16x32_bf16(
                false, a1[kt], false, bfrag[1][kt + 4], (short)0, acc1, false, false);
        }

        // layer 3: per-lane partial dot with W3 over this lane's 2 columns.
        // C layout: row m = r + 8*(lane>>4), column n = lane&15.
        float p[8];
        #pragma unroll
        for (int r = 0; r < 8; ++r)
            p[r] = TANHF(acc0[r]) * w3v0 + TANHF(acc1[r]) * w3v1;

        // reduce over the 16 lanes (n) within each half
        #pragma unroll
        for (int m = 1; m < 16; m <<= 1) {
            #pragma unroll
            for (int r = 0; r < 8; ++r)
                p[r] += __shfl_xor(p[r], m, 32);
        }
        if ((lane & 15) == 0) {
            const int half = lane >> 4;
            #pragma unroll
            for (int r = 0; r < 8; ++r) sRed[wv][half][r] = p[r];
        }
        __syncthreads();   // partials ready

        // Euler update + publish prefetched x (row m = r + 8*half), DT = 1
        if (tid < 16) {
            const int half = tid >> 3, r = tid & 7;
            float s = 0.f;
            #pragma unroll
            for (int w = 0; w < 8; ++w) s += sRed[w][half][r];
            sSt[tid] = sSt[tid] + (s + b3v);
            sX[tid]  = xnext;
        }
        __syncthreads();   // state/x ready for next step
    }
}

extern "C" void kernel_launch(void* const* d_in, const int* in_sizes, int n_in,
                              void* d_out, int out_size, void* d_ws, size_t ws_size,
                              hipStream_t stream) {
    const float* x  = (const float*)d_in[0];
    const float* W1 = (const float*)d_in[1];
    const float* b1 = (const float*)d_in[2];
    const float* W2 = (const float*)d_in[3];
    const float* b2 = (const float*)d_in[4];
    const float* W3 = (const float*)d_in[5];
    const float* b3 = (const float*)d_in[6];
    const float* s0 = (const float*)d_in[7];
    odenet_scan_kernel<<<dim3(BATCH / 16), dim3(256), 0, stream>>>(
        x, W1, b1, W2, b2, W3, b3, s0, (float*)d_out);
}
